// GQAttention_61856118997258
// MI455X (gfx1250) — matrix operand order repaired and checked by reading, not compiled, not run on hardware
//
#include <hip/hip_runtime.h>

// ---------------------------------------------------------------------------
// Types for CDNA5 WMMA
// ---------------------------------------------------------------------------
typedef __bf16 bf16raw;
typedef bf16raw v16bf __attribute__((ext_vector_type(16)));
typedef float   v8f   __attribute__((ext_vector_type(8)));
typedef unsigned uv4 __attribute__((ext_vector_type(4)));
typedef unsigned uv8 __attribute__((ext_vector_type(8)));

union FragU { uint4 q[2]; v16bf v; };
union U8x16 { uint4 q[2]; unsigned short s[16]; };

__device__ inline v16bf make_frag(uint4 a, uint4 b) {
  FragU f; f.q[0] = a; f.q[1] = b; return f.v;
}

__device__ inline unsigned short f2bf(float f) {
  union { float f; unsigned int u; } c; c.f = f;
  unsigned int u = c.u;
  u += 0x7fffu + ((u >> 16) & 1u);     // round-to-nearest-even
  return (unsigned short)(u >> 16);
}
__device__ inline float bf2f(unsigned short h) {
  union { unsigned int u; float f; } c; c.u = ((unsigned int)h) << 16; return c.f;
}

// ---------------------------------------------------------------------------
// Tensor Data Mover: 2D tile load Global -> LDS (issued once per wave).
// D# per CDNA5 ISA: group0 (4 SGPRs) + group1 (8 SGPRs).
// Loads tile_dim1 rows x tile_dim0 elems (2B each), row stride = stride0 elems,
// packed contiguously at lds_addr. Tracked by TENSORcnt.
// ---------------------------------------------------------------------------
__device__ inline void tdm_load_tile_2d(const unsigned short* gptr, unsigned lds_addr,
                                        unsigned tensor_d0, unsigned tensor_d1,
                                        unsigned tile_d0, unsigned tile_d1,
                                        unsigned stride0) {
  unsigned long long ga = (unsigned long long)(uintptr_t)gptr;
  uv4 g0;
  g0[0] = 1u;                                   // count=1, is_restore=0, gather=0
  g0[1] = lds_addr;                             // lds_addr (bits 63:32)
  g0[2] = (unsigned)ga;                         // global_addr lo (bits 95:64)
  g0[3] = ((unsigned)(ga >> 32) & 0x01FFFFFFu)  // global_addr hi (bits 120:96)
          | (2u << 30);                         // type=2 ("image")
  uv8 g1;
  g1[0] = 1u << 16;                             // wg_mask=0, data_size=1 (2B)
  g1[1] = (tensor_d0 & 0xFFFFu) << 16;          // tensor_dim0[15:0] @ bits 63:48
  g1[2] = (tensor_d0 >> 16) | ((tensor_d1 & 0xFFFFu) << 16);  // dim0 hi | dim1 lo
  g1[3] = (tensor_d1 >> 16) | (tile_d0 << 16);  // dim1 hi | tile_dim0 @ 127:112
  g1[4] = tile_d1;                              // tile_dim1 @ 143:128, tile_dim2=0
  g1[5] = stride0;                              // tensor_dim0_stride[31:0]
  g1[6] = 0u;                                   // stride0 hi | dim1_stride lo
  g1[7] = 0u;
  asm volatile("tensor_load_to_lds %0, %1" :: "s"(g0), "s"(g1) : "memory");
}

// ---------------------------------------------------------------------------
// fp32 -> bf16 cast
// ---------------------------------------------------------------------------
__global__ void cast_f32_to_bf16(const float* __restrict__ in,
                                 unsigned short* __restrict__ out, int n) {
  int i = blockIdx.x * blockDim.x + threadIdx.x;
  int stride = gridDim.x * blockDim.x;
  for (; i < n; i += stride) out[i] = f2bf(in[i]);
}

// ---------------------------------------------------------------------------
// Tiled bf16 WMMA GEMM: C[M,N] = A[M,K] @ B[K,N]
// A tile staged by the Tensor Data Mover (wave 0), double-buffered LDS.
// MODE 0: fp32 row-major out; MODE 1: bf16 row-major out;
// MODE 2: bf16 out in V-transposed layout [b][col][s].
// Requires M%128==0, N%128==0, K%32==0 (true for all uses here).
// ---------------------------------------------------------------------------
template <int MODE>
__global__ __launch_bounds__(256) void gemm_bf16_wmma(
    const unsigned short* __restrict__ A,
    const unsigned short* __restrict__ Bm,
    void* __restrict__ Cout,
    int M, int N, int K, int S) {
  __shared__ unsigned short As[2][128][32];   // [buf][m][k]  (TDM destination)
  __shared__ unsigned short Bs[2][128][40];   // [buf][n][k]  (transposed, 16B rows)

  const int t    = threadIdx.x;
  const int wave = t >> 5, lane = t & 31;
  const int lm   = lane & 15, kh = lane >> 4;
  const int wm   = wave >> 1, wn = wave & 1;       // 4 x 2 wave grid
  const unsigned m0 = blockIdx.y * 128u;
  const unsigned n0 = blockIdx.x * 128u;
  const unsigned uK = (unsigned)K, uN = (unsigned)N;

  const unsigned ldsA0 = (unsigned)(uintptr_t)&As[0][0][0];
  const unsigned ldsA1 = (unsigned)(uintptr_t)&As[1][0][0];

  const v8f zero8 = {0.f, 0.f, 0.f, 0.f, 0.f, 0.f, 0.f, 0.f};
  v8f acc[2][4];
#pragma unroll
  for (int i = 0; i < 2; i++)
#pragma unroll
    for (int j = 0; j < 4; j++) acc[i][j] = zero8;

  const int bk = t >> 3, bn = (t & 7) * 16;  // B staging: k-row, 16-col strip
  const int NK = K / 32;

  // ---- prologue: stage buffer 0 for k=0 ----
  if (wave == 0)
    tdm_load_tile_2d(A + (unsigned)(m0 * uK), ldsA0, uK, 128u, 32u, 128u, uK);
  {
    U8x16 bt;
    const uint4* bp = (const uint4*)(Bm + (unsigned)(bk * uN) + n0 + bn);
    bt.q[0] = bp[0]; bt.q[1] = bp[1];
#pragma unroll
    for (int i = 0; i < 16; i++) Bs[0][bn + i][bk] = bt.s[i];
  }
  if (wave == 0) __builtin_amdgcn_s_wait_tensorcnt(0);
  __syncthreads();

  for (int kt = 0; kt < NK; kt++) {
    const int cur = kt & 1;
    const bool have_next = (kt + 1 < NK);
    U8x16 bt;
    // ---- issue next-tile movement before computing on the current tile ----
    if (have_next) {
      const unsigned knext = (unsigned)(kt + 1) * 32u;
      if (wave == 0)
        tdm_load_tile_2d(A + (unsigned)(m0 * uK) + knext,
                         cur ? ldsA0 : ldsA1, uK, 128u, 32u, 128u, uK);
      const uint4* bp = (const uint4*)(Bm + (knext + bk) * uN + n0 + bn);
      bt.q[0] = bp[0]; bt.q[1] = bp[1];
    }

    // ---- fragments from current buffer ----
    v16bf af[2], bfv[4];
#pragma unroll
    for (int tm = 0; tm < 2; tm++) {
      const uint4* p = (const uint4*)&As[cur][wm * 32 + tm * 16 + lm][0];
      af[tm] = make_frag(p[kh], p[2 + kh]);          // K = kh*8+{0..7}, 16+kh*8+{0..7}
    }
#pragma unroll
    for (int tn = 0; tn < 4; tn++) {
      const uint4* p = (const uint4*)&Bs[cur][wn * 64 + tn * 16 + lm][0];
      bfv[tn] = make_frag(p[kh], p[2 + kh]);         // K = kh*8+{0..7}, 16+kh*8+{0..7}
    }
#pragma unroll
    for (int tm = 0; tm < 2; tm++)
#pragma unroll
      for (int tn = 0; tn < 4; tn++)
        acc[tm][tn] = __builtin_amdgcn_wmma_f32_16x16x32_bf16(
            false, af[tm], false, bfv[tn], (short)0, acc[tm][tn], false, false);

    // ---- finish staging next buffer, then rotate ----
    if (have_next) {
#pragma unroll
      for (int i = 0; i < 16; i++) Bs[1 - cur][bn + i][bk] = bt.s[i];
    }
    if (wave == 0) __builtin_amdgcn_s_wait_tensorcnt(0);
    __syncthreads();
  }

  // ---- epilogue ----
  unsigned bb = 0, s0 = 0;
  if (MODE == 2) { bb = m0 / (unsigned)S; s0 = m0 - bb * (unsigned)S; }
#pragma unroll
  for (int tm = 0; tm < 2; tm++) {
#pragma unroll
    for (int tn = 0; tn < 4; tn++) {
#pragma unroll
      for (int r = 0; r < 8; r++) {
        const unsigned row = m0 + wm * 32 + tm * 16 + r + 8 * kh;
        const unsigned col = n0 + wn * 64 + tn * 16 + lm;
        const float v = acc[tm][tn][r];
        if (MODE == 0) {
          ((float*)Cout)[row * uN + col] = v;
        } else if (MODE == 1) {
          ((unsigned short*)Cout)[row * uN + col] = f2bf(v);
        } else {
          const unsigned srow = s0 + wm * 32 + tm * 16 + r + 8 * kh;
          ((unsigned short*)Cout)[(bb * uN + col) * (unsigned)S + srow] = f2bf(v);
        }
      }
    }
  }
}

// ---------------------------------------------------------------------------
// Fused RMSNorm (over head_dim=64) + RoPE (rotate-half), in place on bf16.
// One wave per (token, head); lane owns dims {lane, lane+32}.
// ---------------------------------------------------------------------------
__global__ __launch_bounds__(256) void rmsnorm_rope_kernel(
    unsigned short* __restrict__ buf, const float* __restrict__ w,
    const float* __restrict__ cosT, const float* __restrict__ sinT,
    int rows, int H, int S) {
  const int wave = threadIdx.x >> 5, lane = threadIdx.x & 31;
  const int gid = blockIdx.x * 8 + wave;
  if (gid >= rows * H) return;
  const int token = gid / H;
  const int h = gid - token * H;
  const int s = token % S;
  unsigned short* p = buf + (unsigned)token * (unsigned)(H * 64) + h * 64;
  float e0 = bf2f(p[lane]);
  float e1 = bf2f(p[lane + 32]);
  float ss = e0 * e0 + e1 * e1;
#pragma unroll
  for (int msk = 16; msk >= 1; msk >>= 1) ss += __shfl_xor(ss, msk, 32);
  const float inv = rsqrtf(ss * (1.0f / 64.0f) + 1e-6f);
  const float n0 = e0 * inv * w[lane];
  const float n1 = e1 * inv * w[lane + 32];
  const float c0 = cosT[s * 64 + lane], c1 = cosT[s * 64 + lane + 32];
  const float s0 = sinT[s * 64 + lane], s1 = sinT[s * 64 + lane + 32];
  p[lane]      = f2bf(n0 * c0 - n1 * s0);   // rot(d<32)  = -t[d+32]
  p[lane + 32] = f2bf(n1 * c1 + n0 * s1);   // rot(d>=32) =  t[d-32]
}

// ---------------------------------------------------------------------------
// Causal flash attention (GQA). One wave per (b, h, 16-query tile).
// 32 keys/iter: 4 WMMAs for Q.K^T (K-dim 64) + 4 WMMAs for P.V.
// Causal masking applied only in the (single) diagonal step; softmax runs in
// the log2 domain so each exponential is a bare v_exp_f32.
// ---------------------------------------------------------------------------
__global__ __launch_bounds__(256) void flash_attn_wmma(
    const unsigned short* __restrict__ Qb,
    const unsigned short* __restrict__ Kb,
    const unsigned short* __restrict__ Vt,
    unsigned short* __restrict__ Ob,
    int Bsz, int S, int H, int NKV) {
  __shared__ unsigned short Pl[8][512];   // per-wave P staging, 16x32 bf16

  const int t = threadIdx.x;
  const int wave = t >> 5, lane = t & 31;
  const int lm = lane & 15, kh = lane >> 4;
  const int qtiles = S >> 4;
  const int gid = blockIdx.x * 8 + wave;
  const int qt = gid % qtiles;
  const int tmp = gid / qtiles;
  const int h = tmp % H;
  const int b = tmp / H;
  const int qbase = qt * 16;
  const int kvh = h / (H / NKV);
  const unsigned qstride = (unsigned)(H * 64);
  const unsigned kstride = (unsigned)(NKV * 64);
  const unsigned uS = (unsigned)S;

  const unsigned short* Qp = Qb + (unsigned)(b * S + qbase) * qstride + h * 64;
  const unsigned short* Kp = Kb + (unsigned)(b * S) * kstride + kvh * 64;
  const unsigned short* Vp = Vt + (unsigned)(b * NKV + kvh) * 64u * uS;

  // Q A-fragments (rows qbase..qbase+15, K-dim 64 -> 2 frags), kept in regs.
  v16bf qf[2];
#pragma unroll
  for (int ka = 0; ka < 2; ka++) {
    const unsigned short* p = Qp + (unsigned)lm * qstride + ka * 32;
    uint4 x0 = *(const uint4*)(p + kh * 8);
    uint4 x1 = *(const uint4*)(p + 16 + kh * 8);
    qf[ka] = make_frag(x0, x1);
  }

  const v8f zero8 = {0.f, 0.f, 0.f, 0.f, 0.f, 0.f, 0.f, 0.f};
  v8f o[4];
#pragma unroll
  for (int i = 0; i < 4; i++) o[i] = zero8;
  float mrow[8], lrow[8];
#pragma unroll
  for (int r = 0; r < 8; r++) { mrow[r] = -3.0e38f; lrow[r] = 0.f; }

  // softmax in log2 domain: score * (1/sqrt(64)) * log2(e)
  const float sc2 = 0.125f * 1.44269504088896340736f;
  const int iters = qbase / 32 + 1;      // causal: keys 0 .. qbase+15

  auto kv_step = [&](unsigned jbase, bool masked) {
    // prefetch next key block (speculative; OOB prefetches are dropped)
    __builtin_prefetch(Kp + (jbase + 32u + (unsigned)lane) * kstride, 0, 1);

    // ---- V fragments for this block: issue loads early to overlap with QK^T ----
    uint4 vx[4][2];
#pragma unroll
    for (int ti = 0; ti < 4; ti++) {
      const unsigned short* vrow = Vp + (unsigned)(ti * 16 + lm) * uS + jbase + kh * 8;
      vx[ti][0] = *(const uint4*)vrow;
      vx[ti][1] = *(const uint4*)(vrow + 16);
    }

    // ---- scores: two 16x16 tiles, each chained over K-dim 64 ----
    v8f sfr[2];
#pragma unroll
    for (int st = 0; st < 2; st++) {
      v8f s = zero8;
      const unsigned short* krow = Kp + (jbase + st * 16 + lm) * kstride;
#pragma unroll
      for (int ka = 0; ka < 2; ka++) {
        const unsigned short* p = krow + ka * 32 + kh * 8;
        uint4 x0 = *(const uint4*)p;
        uint4 x1 = *(const uint4*)(p + 16);
        s = __builtin_amdgcn_wmma_f32_16x16x32_bf16(
            false, qf[ka], false, make_frag(x0, x1), (short)0, s, false, false);
      }
      sfr[st] = s;
    }

    // ---- online softmax over the 32 new keys ----
    float alphas[8];
#pragma unroll
    for (int r = 0; r < 8; r++) {
      float v0 = sfr[0][r] * sc2;
      float v1 = sfr[1][r] * sc2;
      if (masked) {   // only the diagonal step pays for the compares
        const int qrow = qbase + r + 8 * kh;
        const int key0 = (int)jbase + lm, key1 = (int)jbase + 16 + lm;
        v0 = (key0 <= qrow) ? v0 : -1.0e30f;
        v1 = (key1 <= qrow) ? v1 : -1.0e30f;
      }
      float mx = fmaxf(v0, v1);
#pragma unroll
      for (int msk = 1; msk < 16; msk <<= 1) mx = fmaxf(mx, __shfl_xor(mx, msk, 32));
      const float mnew = fmaxf(mrow[r], mx);
      const float alpha = exp2f(mrow[r] - mnew);
      const float p0 = exp2f(v0 - mnew);
      const float p1 = exp2f(v1 - mnew);
      float ls = p0 + p1;
#pragma unroll
      for (int msk = 1; msk < 16; msk <<= 1) ls += __shfl_xor(ls, msk, 32);
      lrow[r] = lrow[r] * alpha + ls;
      mrow[r] = mnew;
      alphas[r] = alpha;
      const int prow = (r + 8 * kh) * 32;
      Pl[wave][prow + lm]      = f2bf(p0);
      Pl[wave][prow + 16 + lm] = f2bf(p1);
    }
    // rescale running accumulator (same C-layout row mapping as scores)
#pragma unroll
    for (int ti = 0; ti < 4; ti++)
#pragma unroll
      for (int r = 0; r < 8; r++) o[ti][r] *= alphas[r];

    // wave-local LDS fence: P stores above -> A-frag loads below
    __builtin_amdgcn_wave_barrier();
    asm volatile("s_wait_dscnt 0" ::: "memory");

    // ---- P (16x32) A-fragment from LDS, then P.V over 4 d-tiles ----
    const uint4* pp = (const uint4*)&Pl[wave][lm * 32];
    v16bf pf = make_frag(pp[kh], pp[2 + kh]);
#pragma unroll
    for (int ti = 0; ti < 4; ti++) {
      o[ti] = __builtin_amdgcn_wmma_f32_16x16x32_bf16(
          false, pf, false, make_frag(vx[ti][0], vx[ti][1]), (short)0, o[ti], false, false);
    }
    __builtin_amdgcn_wave_barrier();
  };

  for (int it = 0; it < iters - 1; ++it) kv_step((unsigned)it * 32u, false);
  kv_step((unsigned)(iters - 1) * 32u, true);   // diagonal block

  // ---- normalize and store attention output (bf16, token-major) ----
#pragma unroll
  for (int ti = 0; ti < 4; ti++) {
#pragma unroll
    for (int r = 0; r < 8; r++) {
      const float v = o[ti][r] / lrow[r];
      const unsigned idx =
          (unsigned)(b * S + qbase + r + 8 * kh) * qstride + h * 64 + ti * 16 + lm;
      Ob[idx] = f2bf(v);
    }
  }
}

// ---------------------------------------------------------------------------
// Host launcher
// ---------------------------------------------------------------------------
extern "C" void kernel_launch(void* const* d_in, const int* in_sizes, int n_in,
                              void* d_out, int out_size, void* d_ws, size_t ws_size,
                              hipStream_t stream) {
  (void)in_sizes; (void)n_in; (void)out_size; (void)ws_size;
  constexpr int B = 2, S = 2048, DIN = 2048, H = 32, D = 64, NKV = 8;
  constexpr int M = B * S;

  const float* x    = (const float*)d_in[0];
  // d_in[1] = additive causal mask (handled analytically)
  const float* cosT = (const float*)d_in[2];
  const float* sinT = (const float*)d_in[3];
  const float* Wq   = (const float*)d_in[4];
  const float* Wk   = (const float*)d_in[5];
  const float* Wv   = (const float*)d_in[6];
  const float* Wo   = (const float*)d_in[7];
  const float* qsc  = (const float*)d_in[8];
  const float* ksc  = (const float*)d_in[9];

  char* w = (char*)d_ws;
  size_t off = 0;
  auto alloc = [&](size_t bytes) -> void* {
    void* p = w + off;
    off += (bytes + 255) & ~(size_t)255;
    return p;
  };
  unsigned short* xb  = (unsigned short*)alloc((size_t)M * DIN * 2);
  unsigned short* wqb = (unsigned short*)alloc((size_t)DIN * H * D * 2);
  unsigned short* wkb = (unsigned short*)alloc((size_t)DIN * NKV * D * 2);
  unsigned short* wvb = (unsigned short*)alloc((size_t)DIN * NKV * D * 2);
  unsigned short* wob = (unsigned short*)alloc((size_t)H * D * DIN * 2);
  unsigned short* Qb  = (unsigned short*)alloc((size_t)M * H * D * 2);
  unsigned short* Kb  = (unsigned short*)alloc((size_t)M * NKV * D * 2);
  unsigned short* Vt  = (unsigned short*)alloc((size_t)M * NKV * D * 2);
  unsigned short* Ab  = (unsigned short*)alloc((size_t)M * H * D * 2);

  cast_f32_to_bf16<<<2048, 256, 0, stream>>>(x,  xb,  M * DIN);
  cast_f32_to_bf16<<<2048, 256, 0, stream>>>(Wq, wqb, DIN * H * D);
  cast_f32_to_bf16<<<1024, 256, 0, stream>>>(Wk, wkb, DIN * NKV * D);
  cast_f32_to_bf16<<<1024, 256, 0, stream>>>(Wv, wvb, DIN * NKV * D);
  cast_f32_to_bf16<<<2048, 256, 0, stream>>>(Wo, wob, H * D * DIN);

  dim3 blk(256);
  // Q = x @ Wq  (bf16 out, token-major)
  gemm_bf16_wmma<1><<<dim3((H * D) / 128, M / 128), blk, 0, stream>>>(
      xb, wqb, Qb, M, H * D, DIN, S);
  // K = x @ Wk
  gemm_bf16_wmma<1><<<dim3((NKV * D) / 128, M / 128), blk, 0, stream>>>(
      xb, wkb, Kb, M, NKV * D, DIN, S);
  // V = x @ Wv, written transposed as [b][kvh*64+d][s]
  gemm_bf16_wmma<2><<<dim3((NKV * D) / 128, M / 128), blk, 0, stream>>>(
      xb, wvb, Vt, M, NKV * D, DIN, S);

  rmsnorm_rope_kernel<<<(M * H) / 8,   256, 0, stream>>>(Qb, qsc, cosT, sinT, M, H, S);
  rmsnorm_rope_kernel<<<(M * NKV) / 8, 256, 0, stream>>>(Kb, ksc, cosT, sinT, M, NKV, S);

  flash_attn_wmma<<<(B * H * (S / 16)) / 8, 256, 0, stream>>>(
      Qb, Kb, Vt, Ab, B, S, H, NKV);

  // out = attn @ Wo  (fp32 out)
  gemm_bf16_wmma<0><<<dim3(DIN / 128, M / 128), blk, 0, stream>>>(
      Ab, wob, (float*)d_out, M, DIN, H * D, S);
}
